// HRCMultiheadAttention_57011395887369
// MI455X (gfx1250) — matrix operand order repaired and checked by reading, not compiled, not run on hardware
//
#include <hip/hip_runtime.h>
#include <hip/hip_bf16.h>

// ---------------------------------------------------------------------------
// HRC (random-feature) multi-head attention for MI455X (gfx1250).
// B=4, N=2048, D=1024, H=16, Dh=64, M=256.
//  - All contractions via v_wmma_f32_16x16x32_f16 (f32 accumulate).
//  - phi feature maps regenerated in LDS (never touch HBM).
//  - q/k/v/attn + omega + weights kept as f16 in workspace -> staging is pure
//    16B copies, done with async global->LDS (ASYNCcnt) when available.
//  - All LDS tiles padded (ld=72 / ld=40 halves) for conflict-free b128 frags.
// ---------------------------------------------------------------------------

#define D_MODEL   1024
#define NUM_HEADS 16
#define HEAD_DIM  64
#define M_FEAT    256
#define BATCH     4
#define NSEQ      2048
#define ROWS      (BATCH * NSEQ)          // 8192

#define LD64 72                            // padded ld for 64-wide tiles
#define LD32 40                            // padded ld for 32-wide tiles

typedef __attribute__((ext_vector_type(16))) _Float16 v16h;
typedef __attribute__((ext_vector_type(8)))  float    v8f;
typedef __attribute__((ext_vector_type(4)))  int      v4i;

// --- workspace layout (byte offsets) ---------------------------------------
#define OFF_Q16    ((size_t)0)                    // B*H*N*Dh halves = 16MB
#define OFF_K16    ((size_t)16777216)
#define OFF_V16    ((size_t)33554432)
#define OFF_ATTN16 ((size_t)50331648)             // B*N*C halves
#define OFF_OM16   ((size_t)67108864)             // H*M*Dh halves = 512KB
#define OFF_WQ16   ((size_t)67633152)             // 1024*1024 halves = 2MB
#define OFF_WK16   ((size_t)69730304)
#define OFF_WV16   ((size_t)71827456)
#define OFF_WO16   ((size_t)73924608)
#define OFF_KVC    ((size_t)76021760)             // f32, 4MB
#define OFF_KVS    ((size_t)80216064)
#define OFF_KSC    ((size_t)84410368)             // f32, 64KB
#define OFF_KSS    ((size_t)84475904)
#define ZERO_CNT   (2 * 1048576 + 2 * 16384)      // floats from OFF_KVC

// --- async global->LDS (hedged) --------------------------------------------
#if defined(__AMDGCN__) &&                                              \
    __has_builtin(__builtin_amdgcn_global_load_async_to_lds_b128) &&    \
    __has_builtin(__builtin_amdgcn_s_wait_asynccnt)
#define HAVE_ASYNC_LDS 1
#else
#define HAVE_ASYNC_LDS 0
#endif

__device__ __forceinline__ void async_copy16(const _Float16* g, _Float16* l) {
#if HAVE_ASYNC_LDS
  __builtin_amdgcn_global_load_async_to_lds_b128(
      (__attribute__((address_space(1))) v4i*)g,
      (__attribute__((address_space(3))) v4i*)l, 0, 0);
#else
  *(uint4*)(void*)l = *(const uint4*)(const void*)g;
#endif
}

// wait for this wave's async copies, then workgroup barrier
__device__ __forceinline__ void async_join() {
#if HAVE_ASYNC_LDS
  __builtin_amdgcn_s_wait_asynccnt(0);
#endif
  __syncthreads();
}

// ---------------------------------------------------------------------------
__device__ __forceinline__ v8f vzero8() {
  v8f z;
#pragma unroll
  for (int i = 0; i < 8; ++i) z[i] = 0.0f;
  return z;
}

__device__ __forceinline__ v8f wmma_f16(v16h a, v16h b, v8f c) {
  return __builtin_amdgcn_wmma_f32_16x16x32_f16(false, a, false, b,
                                                (short)0, c, false, false);
}

// --- wave32 fragment loaders (ISA 7.12.2 layouts), contiguous-only ----------
// A 16x32: lane<16 -> row lane&15, elems 0..7=K0..7, 8..15=K16..23;
//          lane>=16 same row, K offset +8.
__device__ __forceinline__ v16h frag_a_row(const _Float16* p, int ld) {
  const int lane = threadIdx.x & 31;
  const int row  = lane & 15;
  const int kb   = (lane & 16) ? 8 : 0;
  const _Float16* q = p + row * ld + kb;
  v16h r;
#pragma unroll
  for (int i = 0; i < 8; ++i) r[i] = q[i];
#pragma unroll
  for (int i = 0; i < 8; ++i) r[8 + i] = q[16 + i];
  return r;
}

// B 32x16: lane<16 -> col lane&15, elems 0..7=K0..7, 8..15=K16..23; lane>=16 K offset +8.
// Source stored as N x K row-major: B[k][col] = p[col*ld + k].
__device__ __forceinline__ v16h frag_b_row(const _Float16* p, int ld) {
  const int lane = threadIdx.x & 31;
  const int col  = lane & 15;
  const int kb   = (lane & 16) ? 8 : 0;
  const _Float16* q = p + col * ld + kb;
  v16h r;
#pragma unroll
  for (int i = 0; i < 8; ++i) r[i] = q[i];
#pragma unroll
  for (int i = 0; i < 8; ++i) r[8 + i] = q[16 + i];
  return r;
}

// --- staging ---------------------------------------------------------------
// Pure f16 copy of a 64 x W tile (srcLd halves) into LDS with dst ld = DLD.
template <int W, int DLD>
__device__ __forceinline__ void stage_copy(const _Float16* __restrict__ src,
                                           int srcLd, _Float16* dst) {
  const int t = threadIdx.x;
  constexpr int CPR = W / 8;               // 16B chunks per row
  constexpr int ITERS = (64 * CPR) / 256;
#pragma unroll
  for (int j = 0; j < ITERS; ++j) {
    const int idx = t + j * 256;
    const int row = idx / CPR;
    const int c   = (idx % CPR) * 8;
    async_copy16(src + row * srcLd + c, dst + row * DLD + c);
  }
}

// f32 -> f16 convert of a 64x32 tile into LDS (ld = LD32)
__device__ __forceinline__ void stage_cvt_64x32(const float* __restrict__ src,
                                                int srcLd, _Float16* dst) {
  const int t = threadIdx.x;
#pragma unroll
  for (int j = 0; j < 2; ++j) {
    const int idx = t + j * 256;           // 512 float4
    const int row = idx >> 3;
    const int c   = (idx & 7) << 2;
    const float4 val = *(const float4*)(src + row * srcLd + c);
    _Float16* d = dst + row * LD32 + c;
    d[0] = (_Float16)val.x; d[1] = (_Float16)val.y;
    d[2] = (_Float16)val.z; d[3] = (_Float16)val.w;
  }
}

// overload dispatch for proj_gemm X staging (must precede the template)
__device__ __forceinline__ void stage_cvt_or_copy(const float* src,
                                                  _Float16* dst) {
  stage_cvt_64x32(src, D_MODEL, dst);
}
__device__ __forceinline__ void stage_cvt_or_copy(const _Float16* src,
                                                  _Float16* dst) {
  stage_copy<32, LD32>(src, D_MODEL, dst);
}

// f16 64x64 tile -> transposed LDS tile dst[d][n], ld = LD64
__device__ __forceinline__ void stage_vT(const _Float16* __restrict__ src,
                                         int srcLd, _Float16* dst) {
  const int t = threadIdx.x;
#pragma unroll
  for (int j = 0; j < 2; ++j) {
    const int idx = t + j * 256;           // 512 chunks of 8 halves
    const int row = idx >> 3;              // n
    const int c0  = (idx & 7) << 3;        // d base
    uint4 raw = *(const uint4*)(src + row * srcLd + c0);
    const _Float16* h = (const _Float16*)&raw;
#pragma unroll
    for (int e = 0; e < 8; ++e) dst[(c0 + e) * LD64 + row] = h[e];
  }
}

// f32 64x64 tile (row-major m x d) -> f16 transposed LDS dst[d][m], ld = LD64
__device__ __forceinline__ void stage_kvT(const float* __restrict__ src,
                                          _Float16* dst) {
  const int t = threadIdx.x;
#pragma unroll
  for (int j = 0; j < 4; ++j) {
    const int idx = t + j * 256;           // 1024 float4
    const int row = idx >> 4;              // m
    const int c   = (idx & 15) << 2;       // d
    const float4 val = *(const float4*)(src + row * 64 + c);
    dst[(c + 0) * LD64 + row] = (_Float16)val.x;
    dst[(c + 1) * LD64 + row] = (_Float16)val.y;
    dst[(c + 2) * LD64 + row] = (_Float16)val.z;
    dst[(c + 3) * LD64 + row] = (_Float16)val.w;
  }
}

// ---------------------------------------------------------------------------
// out(8192 x 1024) = X @ W^T + bias (optional scale). W pre-converted to f16.
// HEAD_OUT: f16 (B,H,N,Dh) output; else TOUT row-major (B,N,C).
// ---------------------------------------------------------------------------
template <typename TIN, typename TOUT, bool HEAD_OUT>
__global__ __launch_bounds__(256) void proj_gemm(
    const TIN* __restrict__ X, const _Float16* __restrict__ W16,
    const float* __restrict__ bias, TOUT* __restrict__ out, float scale) {
  __shared__ _Float16 x_lds[64 * LD32];
  __shared__ _Float16 w_lds[64 * LD32];
  const int rowbase = blockIdx.y * 64;
  const int colbase = blockIdx.x * 64;
  const int wave = threadIdx.x >> 5;
  const int lane = threadIdx.x & 31;
  const int col  = lane & 15;
  const int r0   = (lane & 16) ? 8 : 0;

  v8f acc[2];
  acc[0] = vzero8(); acc[1] = vzero8();

  for (int kk = 0; kk < D_MODEL; kk += 32) {
    __syncthreads();
    stage_cvt_or_copy(X + (size_t)rowbase * D_MODEL + kk, x_lds);
    stage_copy<32, LD32>(W16 + (size_t)colbase * D_MODEL + kk, D_MODEL, w_lds);
    async_join();
#pragma unroll
    for (int i = 0; i < 2; ++i) {
      const int t = wave * 2 + i;
      const int trow = t >> 2, tcol = t & 3;
      v16h a = frag_a_row(x_lds + trow * 16 * LD32, LD32);
      v16h b = frag_b_row(w_lds + tcol * 16 * LD32, LD32);
      acc[i] = wmma_f16(a, b, acc[i]);
    }
  }

#pragma unroll
  for (int i = 0; i < 2; ++i) {
    const int t = wave * 2 + i;
    const int trow = t >> 2, tcol = t & 3;
    const int cg = colbase + tcol * 16 + col;
    const float bv = bias[cg];
#pragma unroll
    for (int e = 0; e < 8; ++e) {
      const int rg = rowbase + trow * 16 + r0 + e;
      const float val = (acc[i][e] + bv) * scale;
      if (HEAD_OUT) {
        const int bb = rg >> 11, nn = rg & 2047;
        const int hh = cg >> 6,  dd = cg & 63;
        out[((size_t)(bb * NUM_HEADS + hh) * NSEQ + nn) * HEAD_DIM + dd] =
            (TOUT)val;
      } else {
        out[(size_t)rg * D_MODEL + cg] = (TOUT)val;
      }
    }
  }
}

// ---------------------------------------------------------------------------
// kv kernel: per (b,h, 256-row chunk) regenerate phi_k tiles (transposed in
// LDS) and accumulate kv_c/kv_s = phi^T @ v and ksum column sums via atomics.
// ---------------------------------------------------------------------------
__global__ __launch_bounds__(256) void kv_kernel(
    const _Float16* __restrict__ kbuf, const _Float16* __restrict__ vbuf,
    const _Float16* __restrict__ om16,
    float* __restrict__ kv_c, float* __restrict__ kv_s,
    float* __restrict__ ksum_c, float* __restrict__ ksum_s) {
  __shared__ _Float16 k_lds[64 * LD64];    // [n][d]
  __shared__ _Float16 vT_lds[64 * LD64];   // [d][n]
  __shared__ _Float16 om_lds[64 * LD64];   // [m][d]
  __shared__ _Float16 fkcT[64 * LD64];     // [m][n]
  __shared__ _Float16 fksT[64 * LD64];     // [m][n]
  __shared__ float sk_lds[64];

  const int bh    = blockIdx.x;
  const int h     = bh & 15;
  const int chunk = blockIdx.y;
  const int wave  = threadIdx.x >> 5;
  const int lane  = threadIdx.x & 31;
  const int col   = lane & 15;
  const int r0    = (lane & 16) ? 8 : 0;
  const int t0    = wave * 2;

  for (int ms = 0; ms < 4; ++ms) {
    __syncthreads();
    stage_copy<64, LD64>(om16 + (size_t)(h * M_FEAT + ms * 64) * HEAD_DIM,
                         HEAD_DIM, om_lds);
    v8f accC[2], accS[2];
    accC[0] = vzero8(); accC[1] = vzero8();
    accS[0] = vzero8(); accS[1] = vzero8();

    for (int ns = 0; ns < 4; ++ns) {
      const int nbase = chunk * 256 + ns * 64;
      __syncthreads();
      stage_copy<64, LD64>(kbuf + ((size_t)bh * NSEQ + nbase) * HEAD_DIM,
                           HEAD_DIM, k_lds);
      stage_vT(vbuf + ((size_t)bh * NSEQ + nbase) * HEAD_DIM, HEAD_DIM, vT_lds);
      async_join();
      if (threadIdx.x < 64) {
        float s = 0.0f;
        for (int d = 0; d < 64; ++d) {
          const float xv = (float)k_lds[threadIdx.x * LD64 + d];
          s += xv * xv;
        }
        sk_lds[threadIdx.x] = __expf(0.5f * s) * 0.0625f;   // /sqrt(M)
      }
      __syncthreads();

      // theta tiles (4 n-strips x 4 m16-tiles), 2/wave -> transposed phi LDS
#pragma unroll
      for (int i = 0; i < 2; ++i) {
        const int t = t0 + i;
        const int trow = t >> 2, tcol = t & 3;
        v8f th = vzero8();
#pragma unroll
        for (int kkc = 0; kkc < 64; kkc += 32) {
          v16h a = frag_a_row(k_lds + trow * 16 * LD64 + kkc, LD64);
          v16h b = frag_b_row(om_lds + tcol * 16 * LD64 + kkc, LD64);
          th = wmma_f16(a, b, th);
        }
#pragma unroll
        for (int e = 0; e < 8; ++e) {
          const int nl = trow * 16 + r0 + e;
          const int ml = tcol * 16 + col;
          const float s = sk_lds[nl];
          const float ang = th[e];
          fkcT[ml * LD64 + nl] = (_Float16)(s * __cosf(ang));
          fksT[ml * LD64 + nl] = (_Float16)(s * __sinf(ang));
        }
      }
      __syncthreads();

      // ksum rows of transposed phi (contiguous), concurrent with kv WMMA
      if (threadIdx.x < 128) {
        const int m = threadIdx.x & 63;
        const _Float16* f = (threadIdx.x < 64) ? fkcT : fksT;
        float* dst = (threadIdx.x < 64) ? ksum_c : ksum_s;
        float s = 0.0f;
        for (int n = 0; n < 64; ++n) s += (float)f[m * LD64 + n];
        atomicAdd(&dst[(size_t)bh * M_FEAT + ms * 64 + m], s);
      }

      // kv accumulation: A = phi^T (row-major in fk*T), B = v^T (row-major)
#pragma unroll
      for (int i = 0; i < 2; ++i) {
        const int t = t0 + i;
        const int trow = t >> 2, tcol = t & 3;
#pragma unroll
        for (int kkc = 0; kkc < 64; kkc += 32) {
          v16h ac = frag_a_row(fkcT + trow * 16 * LD64 + kkc, LD64);
          v16h as = frag_a_row(fksT + trow * 16 * LD64 + kkc, LD64);
          v16h bb = frag_b_row(vT_lds + tcol * 16 * LD64 + kkc, LD64);
          accC[i] = wmma_f16(ac, bb, accC[i]);
          accS[i] = wmma_f16(as, bb, accS[i]);
        }
      }
    }

#pragma unroll
    for (int i = 0; i < 2; ++i) {
      const int t = t0 + i;
      const int trow = t >> 2, tcol = t & 3;
#pragma unroll
      for (int e = 0; e < 8; ++e) {
        const int m = ms * 64 + trow * 16 + r0 + e;
        const int d = tcol * 16 + col;
        atomicAdd(&kv_c[((size_t)bh * M_FEAT + m) * HEAD_DIM + d], accC[i][e]);
        atomicAdd(&kv_s[((size_t)bh * M_FEAT + m) * HEAD_DIM + d], accS[i][e]);
      }
    }
  }
}

// ---------------------------------------------------------------------------
// numerator kernel: per (b,h, 64-row chunk) regenerate phi_q, numerator =
// phi_q @ kv, denominator = phi_q . ksum, normalize, write f16 activations.
// ---------------------------------------------------------------------------
__global__ __launch_bounds__(256) void num_kernel(
    const _Float16* __restrict__ qbuf, const _Float16* __restrict__ om16,
    const float* __restrict__ kv_c, const float* __restrict__ kv_s,
    const float* __restrict__ ksum_c, const float* __restrict__ ksum_s,
    _Float16* __restrict__ attn) {
  __shared__ _Float16 q_lds[64 * LD64];    // [n][d]
  __shared__ _Float16 om_lds[64 * LD64];   // [m][d]
  __shared__ _Float16 fqc[64 * LD64];      // [n][m]
  __shared__ _Float16 fqs[64 * LD64];      // [n][m]
  __shared__ _Float16 kvcT[64 * LD64];     // [d][m]
  __shared__ _Float16 kvsT[64 * LD64];     // [d][m]
  __shared__ float sq_lds[64];
  __shared__ float den_lds[64];

  const int bh    = blockIdx.x;
  const int h     = bh & 15;
  const int b     = bh >> 4;
  const int nbase = blockIdx.y * 64;
  const int wave  = threadIdx.x >> 5;
  const int lane  = threadIdx.x & 31;
  const int col   = lane & 15;
  const int r0    = (lane & 16) ? 8 : 0;
  const int t0    = wave * 2;

  stage_copy<64, LD64>(qbuf + ((size_t)bh * NSEQ + nbase) * HEAD_DIM, HEAD_DIM,
                       q_lds);
  async_join();
  if (threadIdx.x < 64) {
    float s = 0.0f;
    for (int d = 0; d < 64; ++d) {
      const float xv = (float)q_lds[threadIdx.x * LD64 + d];
      s += xv * xv;
    }
    sq_lds[threadIdx.x]  = __expf(0.5f * s) * 0.0625f;
    den_lds[threadIdx.x] = 0.0f;
  }
  __syncthreads();

  v8f acc[2];
  acc[0] = vzero8(); acc[1] = vzero8();

  for (int ms = 0; ms < 4; ++ms) {
    __syncthreads();
    stage_copy<64, LD64>(om16 + (size_t)(h * M_FEAT + ms * 64) * HEAD_DIM,
                         HEAD_DIM, om_lds);
    stage_kvT(kv_c + ((size_t)bh * M_FEAT + ms * 64) * HEAD_DIM, kvcT);
    stage_kvT(kv_s + ((size_t)bh * M_FEAT + ms * 64) * HEAD_DIM, kvsT);
    async_join();

    // theta_q tiles -> phi_q tiles ([n][m], row-major)
#pragma unroll
    for (int i = 0; i < 2; ++i) {
      const int t = t0 + i;
      const int trow = t >> 2, tcol = t & 3;
      v8f th = vzero8();
#pragma unroll
      for (int kkc = 0; kkc < 64; kkc += 32) {
        v16h a = frag_a_row(q_lds + trow * 16 * LD64 + kkc, LD64);
        v16h bb = frag_b_row(om_lds + tcol * 16 * LD64 + kkc, LD64);
        th = wmma_f16(a, bb, th);
      }
#pragma unroll
      for (int e = 0; e < 8; ++e) {
        const int nl = trow * 16 + r0 + e;
        const float s = sq_lds[nl];
        const float ang = th[e];
        fqc[nl * LD64 + tcol * 16 + col] = (_Float16)(s * __cosf(ang));
        fqs[nl * LD64 + tcol * 16 + col] = (_Float16)(s * __sinf(ang));
      }
    }
    __syncthreads();

    // denominator (row-exclusive)
    if (threadIdx.x < 64) {
      float s = den_lds[threadIdx.x];
      const size_t kb = (size_t)bh * M_FEAT + ms * 64;
      for (int m = 0; m < 64; ++m) {
        s += (float)fqc[threadIdx.x * LD64 + m] * ksum_c[kb + m]
           + (float)fqs[threadIdx.x * LD64 + m] * ksum_s[kb + m];
      }
      den_lds[threadIdx.x] = s;
    }

    // numerator: phi_c @ kv_c + phi_s @ kv_s into same accumulator
#pragma unroll
    for (int i = 0; i < 2; ++i) {
      const int t = t0 + i;
      const int trow = t >> 2, tcol = t & 3;
#pragma unroll
      for (int kkc = 0; kkc < 64; kkc += 32) {
        v16h ac = frag_a_row(fqc + trow * 16 * LD64 + kkc, LD64);
        v16h bc = frag_b_row(kvcT + tcol * 16 * LD64 + kkc, LD64);
        acc[i] = wmma_f16(ac, bc, acc[i]);
        v16h as = frag_a_row(fqs + trow * 16 * LD64 + kkc, LD64);
        v16h bs = frag_b_row(kvsT + tcol * 16 * LD64 + kkc, LD64);
        acc[i] = wmma_f16(as, bs, acc[i]);
      }
    }
  }
  __syncthreads();

#pragma unroll
  for (int i = 0; i < 2; ++i) {
    const int t = t0 + i;
    const int trow = t >> 2, tcol = t & 3;
#pragma unroll
    for (int e = 0; e < 8; ++e) {
      const int nl = trow * 16 + r0 + e;
      const float den = den_lds[nl] + 1e-6f;
      attn[((size_t)(b * NSEQ + nbase + nl)) * D_MODEL + h * HEAD_DIM +
           tcol * 16 + col] = (_Float16)(acc[i][e] / den);
    }
  }
}

// ---------------------------------------------------------------------------
__global__ void cvt_f32_f16(const float* __restrict__ src,
                            _Float16* __restrict__ dst, int n4) {
  const int i = blockIdx.x * 256 + threadIdx.x;   // float4 index
  if (i < n4) {
    const float4 v = *(const float4*)(src + (size_t)i * 4);
    _Float16* d = dst + (size_t)i * 4;
    d[0] = (_Float16)v.x; d[1] = (_Float16)v.y;
    d[2] = (_Float16)v.z; d[3] = (_Float16)v.w;
  }
}

__global__ void zero_kernel(float* __restrict__ p, int n) {
  const int i = blockIdx.x * 256 + threadIdx.x;
  if (i < n) p[i] = 0.0f;
}

// ---------------------------------------------------------------------------
extern "C" void kernel_launch(void* const* d_in, const int* in_sizes, int n_in,
                              void* d_out, int out_size, void* d_ws,
                              size_t ws_size, hipStream_t stream) {
  (void)in_sizes; (void)n_in; (void)out_size; (void)ws_size;
  const float* x     = (const float*)d_in[0];
  const float* Wq    = (const float*)d_in[1];
  const float* bq    = (const float*)d_in[2];
  const float* Wk    = (const float*)d_in[3];
  const float* bk    = (const float*)d_in[4];
  const float* Wv    = (const float*)d_in[5];
  const float* bv    = (const float*)d_in[6];
  const float* Wo    = (const float*)d_in[7];
  const float* bo    = (const float*)d_in[8];
  const float* omega = (const float*)d_in[9];
  float* out = (float*)d_out;
  char* w = (char*)d_ws;

  _Float16* q16    = (_Float16*)(w + OFF_Q16);
  _Float16* k16    = (_Float16*)(w + OFF_K16);
  _Float16* v16    = (_Float16*)(w + OFF_V16);
  _Float16* attn16 = (_Float16*)(w + OFF_ATTN16);
  _Float16* om16   = (_Float16*)(w + OFF_OM16);
  _Float16* wq16   = (_Float16*)(w + OFF_WQ16);
  _Float16* wk16   = (_Float16*)(w + OFF_WK16);
  _Float16* wv16   = (_Float16*)(w + OFF_WV16);
  _Float16* wo16   = (_Float16*)(w + OFF_WO16);
  float* kvc = (float*)(w + OFF_KVC);
  float* kvs = (float*)(w + OFF_KVS);
  float* ksc = (float*)(w + OFF_KSC);
  float* kss = (float*)(w + OFF_KSS);

  const float qk_scale = 0.35355339059327373f;    // 64^-0.25

  dim3 blk(256);
  dim3 gproj(D_MODEL / 64, ROWS / 64);            // (16, 128)

  // pre-convert omega + weights to f16
  cvt_f32_f16<<<(65536 + 255) / 256, blk, 0, stream>>>(omega, om16, 65536);
  cvt_f32_f16<<<(262144 + 255) / 256, blk, 0, stream>>>(Wq, wq16, 262144);
  cvt_f32_f16<<<(262144 + 255) / 256, blk, 0, stream>>>(Wk, wk16, 262144);
  cvt_f32_f16<<<(262144 + 255) / 256, blk, 0, stream>>>(Wv, wv16, 262144);
  cvt_f32_f16<<<(262144 + 255) / 256, blk, 0, stream>>>(Wo, wo16, 262144);

  proj_gemm<float, _Float16, true><<<gproj, blk, 0, stream>>>(
      x, wq16, bq, q16, qk_scale);
  proj_gemm<float, _Float16, true><<<gproj, blk, 0, stream>>>(
      x, wk16, bk, k16, qk_scale);
  proj_gemm<float, _Float16, true><<<gproj, blk, 0, stream>>>(
      x, wv16, bv, v16, 1.0f);

  zero_kernel<<<(ZERO_CNT + 255) / 256, blk, 0, stream>>>(kvc, ZERO_CNT);

  kv_kernel<<<dim3(BATCH * NUM_HEADS, NSEQ / 256), blk, 0, stream>>>(
      k16, v16, om16, kvc, kvs, ksc, kss);
  num_kernel<<<dim3(BATCH * NUM_HEADS, NSEQ / 64), blk, 0, stream>>>(
      q16, om16, kvc, kvs, ksc, kss, attn16);

  proj_gemm<_Float16, float, false><<<gproj, blk, 0, stream>>>(
      attn16, wo16, bo, out, 1.0f);
}
